// NodeAttention_65068754534661
// MI455X (gfx1250) — matrix operand order repaired and checked by reading, not compiled, not run on hardware
//
#include <hip/hip_runtime.h>

typedef __attribute__((ext_vector_type(2)))  float  v2f;
typedef __attribute__((ext_vector_type(8)))  float  v8f;
typedef __attribute__((ext_vector_type(16))) __bf16 v16bf;

#define NEG_INF -9.0e15f
#define SLOPE   0.01f

__device__ __forceinline__ float leaky(float x) { return x > 0.0f ? x : SLOPE * x; }

// round-to-nearest-even f32 -> bf16
__device__ __forceinline__ unsigned short f2bf(float f) {
    unsigned int x = __float_as_uint(f);
    x += 0x7FFFu + ((x >> 16) & 1u);
    return (unsigned short)(x >> 16);
}

union ABf { v16bf v; unsigned int u[8]; uint4 q[2]; };

// ---------------- Kernel 1: h = x @ W  (fp32 WMMA 16x16x4) ----------------
// grid (4, 512), block 128 (4 waves); wave computes one 16x16 tile of h.
__global__ __launch_bounds__(128) void gat_gemm_h(const float* __restrict__ x,
                                                  const float* __restrict__ W,
                                                  float* __restrict__ h) {
    const int lane = threadIdx.x & 31;
    const int wave = threadIdx.x >> 5;
    const int i0 = blockIdx.y * 16;
    const int c0 = (blockIdx.x * 4 + wave) * 16;
    const int m    = lane & 15;
    const int kh   = (lane >> 4) * 2;   // K half-offset per A/B fp32 layout

    v8f acc = {};
    const float* xrow = x + (size_t)(i0 + m) * 512 + kh;
    const float* wcol = W + (size_t)kh * 256 + c0 + m;
    for (int k = 0; k < 512; k += 4) {
        v2f a; a.x = xrow[k];             a.y = xrow[k + 1];               // A 16x4
        v2f b; b.x = wcol[(size_t)k*256]; b.y = wcol[(size_t)k*256 + 256]; // B 4x16
        acc = __builtin_amdgcn_wmma_f32_16x16x4_f32(false, a, false, b,
                                                    (short)0, acc, false, false);
    }
    const int mrow = 8 * (lane >> 4);   // C layout: vgpr r -> row r (+8 upper half)
    #pragma unroll
    for (int r = 0; r < 8; ++r)
        h[(size_t)(i0 + mrow + r) * 256 + c0 + m] = acc[r];
}

// ------------- Kernel 2: f1/f2 = h@a1, h@a2 ; hbfT = bf16(h)^T -------------
// grid 1024, block 256 (8 waves); one wave per row.
__global__ __launch_bounds__(256) void gat_feat(const float* __restrict__ h,
                                                const float* __restrict__ a,
                                                float* __restrict__ f1,
                                                float* __restrict__ f2,
                                                unsigned short* __restrict__ hbfT) {
    const int lane = threadIdx.x & 31;
    const int wave = threadIdx.x >> 5;
    const int i = blockIdx.x * 8 + wave;
    float s1 = 0.0f, s2 = 0.0f;
    #pragma unroll
    for (int q = 0; q < 8; ++q) {
        int c = lane + 32 * q;
        float hv = h[(size_t)i * 256 + c];
        s1 = fmaf(hv, a[c], s1);
        s2 = fmaf(hv, a[256 + c], s2);
        hbfT[(size_t)c * 8192 + i] = f2bf(hv);   // transposed bf16 copy for WMMA B loads
    }
    #pragma unroll
    for (int off = 16; off > 0; off >>= 1) {
        s1 += __shfl_xor(s1, off, 32);
        s2 += __shfl_xor(s2, off, 32);
    }
    if (lane == 0) { f1[i] = s1; f2[i] = s2; }
}

// --------- Kernel 3: fused masked-softmax attention, P@h via bf16 WMMA ---------
// grid 512, block 256 (8 waves). Block owns 16 rows; wave owns 32 output cols.
// 64-column J-tiles: 4 adj elements/thread, 4 WMMAs/wave between each barrier pair.
__global__ __launch_bounds__(256) void gat_attn(const int* __restrict__ adj,
                                                const float* __restrict__ f1,
                                                const float* __restrict__ f2,
                                                const unsigned short* __restrict__ hbfT,
                                                float* __restrict__ out) {
    __shared__ float f1s[16];
    __shared__ float scaleS[16];
    __shared__ float sinvS[16];
    __shared__ unsigned int pk[16][33];          // 32 bf16-pairs/row; stride 33 = conflict-free

    const int tid  = threadIdx.x;
    const int lane = tid & 31;
    const int wave = tid >> 5;
    const int i0   = blockIdx.x * 16;

    if (tid < 16) f1s[tid] = f1[i0 + tid];
    __syncthreads();

    const int irow = tid >> 4;                   // producer row 0..15 (== 2*wave + half)
    const int jp   = tid & 15;                   // producer quad index (4 columns)
    const float f1i = f1s[irow];
    const int* __restrict__ arow = adj + (size_t)(i0 + irow) * 8192 + jp * 4;
    const float* __restrict__ frow = f2 + jp * 4;

    const int n    = lane & 15;
    const int half = lane >> 4;
    const int c0   = wave * 32;
    const size_t bbase0 = (size_t)(c0 + n)      * 8192 + half * 16;
    const size_t bbase1 = (size_t)(c0 + 16 + n) * 8192 + half * 16;

    v8f acc0 = {}, acc1 = {};
    float Mreg = NEG_INF, Sreg = 0.0f;           // live in lanes 0 and 16

    for (int j0 = 0; j0 < 8192; j0 += 64) {
        // prefetch adjacency 2 iterations (128 cols) ahead of the stream
        if (j0 + 128 < 8192) __builtin_prefetch(arow + j0 + 128, 0, 1);

        // ---- scores + online softmax state (shuffle reductions in half-waves) ----
        int4   ad = *(const int4*)(arow + j0);
        float4 fj = *(const float4*)(frow + j0);
        float m0 = ad.x > 0 ? leaky(f1i + fj.x) : NEG_INF;
        float m1 = ad.y > 0 ? leaky(f1i + fj.y) : NEG_INF;
        float m2 = ad.z > 0 ? leaky(f1i + fj.z) : NEG_INF;
        float m3 = ad.w > 0 ? leaky(f1i + fj.w) : NEG_INF;
        float pm = fmaxf(fmaxf(m0, m1), fmaxf(m2, m3));
        pm = fmaxf(pm, __shfl_xor(pm, 1, 32));
        pm = fmaxf(pm, __shfl_xor(pm, 2, 32));
        pm = fmaxf(pm, __shfl_xor(pm, 4, 32));
        pm = fmaxf(pm, __shfl_xor(pm, 8, 32));
        float Mnew  = fmaxf(Mreg, pm);
        float scale = __expf(Mreg - Mnew);       // finite NEG_INF => always well-defined
        float Mn = __shfl(Mnew, half << 4, 32);  // broadcast row max from lane 0/16
        float p0 = __expf(m0 - Mn);
        float p1 = __expf(m1 - Mn);
        float p2 = __expf(m2 - Mn);
        float p3 = __expf(m3 - Mn);
        float ps = (p0 + p1) + (p2 + p3);
        ps += __shfl_xor(ps, 1, 32);
        ps += __shfl_xor(ps, 2, 32);
        ps += __shfl_xor(ps, 4, 32);
        ps += __shfl_xor(ps, 8, 32);
        if ((lane & 15) == 0) {
            Mreg = Mnew;
            Sreg = Sreg * scale + ps;
            scaleS[irow] = scale;
        }
        pk[irow][2 * jp]     = (unsigned int)f2bf(p0) | ((unsigned int)f2bf(p1) << 16);
        pk[irow][2 * jp + 1] = (unsigned int)f2bf(p2) | ((unsigned int)f2bf(p3) << 16);
        __syncthreads();

        // ---- one rescale per 64 cols, then 2 K-chunks x 2 N-tiles of WMMA ----
        #pragma unroll
        for (int r = 0; r < 8; ++r) {
            float sc = scaleS[r + 8 * half];
            acc0[r] *= sc; acc1[r] *= sc;
        }
        #pragma unroll
        for (int chunk = 0; chunk < 2; ++chunk) {
            ABf A;
            #pragma unroll
            for (int v = 0; v < 8; ++v) {
                int kp = v + 8 * half;  // contiguous K order (k = 16*half + s), matches B loads
                A.u[v] = pk[n][chunk * 16 + kp];
            }
            ABf B0, B1;
            const uint4* hb0 = (const uint4*)(hbfT + bbase0 + j0 + chunk * 32);
            const uint4* hb1 = (const uint4*)(hbfT + bbase1 + j0 + chunk * 32);
            B0.q[0] = hb0[0]; B0.q[1] = hb0[1];
            B1.q[0] = hb1[0]; B1.q[1] = hb1[1];
            acc0 = __builtin_amdgcn_wmma_f32_16x16x32_bf16(false, A.v, false, B0.v,
                                                           (short)0, acc0, false, false);
            acc1 = __builtin_amdgcn_wmma_f32_16x16x32_bf16(false, A.v, false, B1.v,
                                                           (short)0, acc1, false, false);
        }
        __syncthreads();
    }

    if ((lane & 15) == 0) sinvS[wave * 2 + half] = 1.0f / Sreg;
    __syncthreads();
    #pragma unroll
    for (int r = 0; r < 8; ++r) {
        float si = sinvS[r + 8 * half];
        size_t row = (size_t)(i0 + r + 8 * half);
        out[row * 256 + c0 + n]      = leaky(acc0[r] * si);
        out[row * 256 + c0 + 16 + n] = leaky(acc1[r] * si);
    }
}

extern "C" void kernel_launch(void* const* d_in, const int* in_sizes, int n_in,
                              void* d_out, int out_size, void* d_ws, size_t ws_size,
                              hipStream_t stream) {
    const float* x   = (const float*)d_in[0];
    const int*   adj = (const int*)d_in[1];
    const float* W   = (const float*)d_in[2];
    const float* a   = (const float*)d_in[3];
    float* out = (float*)d_out;

    char* ws = (char*)d_ws;
    float*          h    = (float*)ws;                                   // 8 MB
    unsigned short* hbfT = (unsigned short*)(ws + (size_t)8192*256*4);   // 4 MB
    float*          f1   = (float*)(ws + (size_t)8192*256*6);            // 32 KB
    float*          f2   = f1 + 8192;                                    // 32 KB

    gat_gemm_h<<<dim3(4, 512), 128, 0, stream>>>(x, W, h);
    gat_feat  <<<1024, 256, 0, stream>>>(h, a, f1, f2, hbfT);
    gat_attn  <<<512, 256, 0, stream>>>(adj, f1, f2, hbfT, out);
}
